// Attention_61443802137307
// MI455X (gfx1250) — matrix operand order repaired and checked by reading, not compiled, not run on hardware
//
#include <hip/hip_runtime.h>

// ---------------- problem constants ----------------
#define DIMX   512
#define HEADS  8
#define HD     64
#define SEQ    144
#define SEQP   160          // padded KV length for WMMA K-loop (multiple of 32)
#define BATCH  256
#define NTOK   (BATCH * SEQ)        // 36864 tokens
#define MBLOCKS (NTOK / 128)        // 288: 8 waves x 16 rows per block

typedef __attribute__((ext_vector_type(16))) _Float16 v16h;
typedef __attribute__((ext_vector_type(8)))  _Float16 v8h;
typedef __attribute__((ext_vector_type(4)))  _Float16 v4h;
typedef __attribute__((ext_vector_type(8)))  float    v8f;

union V16U { v16h v; v8h h[2]; };

// ---- WMMA fragment loaders (CDNA5 16x16x32 f16 layouts, ISA 7.12.2) ----
// A (16x32, MxK): lane<16 holds K {0..7,16..23}, lane>=16 holds K {8..15,24..31}
static __device__ inline v16h load_a_frag(const _Float16* __restrict__ row, int kblk, int lane) {
    const int lo = (lane & 16) ? 8 : 0;
    V16U u;
    u.h[0] = *(const v8h*)(row + kblk * 32 + lo);
    u.h[1] = *(const v8h*)(row + kblk * 32 + 16 + lo);
    return u.v;
}
// B (32x16, KxN): lane<16 holds K {0..7,16..23}, lane>=16 holds K {8..15,24..31} (fixed col = lane&15)
static __device__ inline v16h load_b_frag(const _Float16* __restrict__ colrow, int kblk, int lane) {
    const int kb = (lane & 16) ? 8 : 0;
    const _Float16* p = colrow + kblk * 32 + kb;
    V16U u;
    u.h[0] = *(const v8h*)(p);
    u.h[1] = *(const v8h*)(p + 16);
    return u.v;
}

static __device__ inline v8f wmma_f16(v16h a, v16h b, v8f c) {
    return __builtin_amdgcn_wmma_f32_16x16x32_f16(false, a, false, b, (short)0, c, false, false);
}

// ---------------- conversion kernels ----------------
__global__ __launch_bounds__(256) void cvt_x_kernel(const float* __restrict__ x,
                                                    const float* __restrict__ topo,
                                                    const int* __restrict__ is_end,
                                                    _Float16* __restrict__ x16,
                                                    _Float16* __restrict__ xt16) {
    size_t i = (size_t)blockIdx.x * blockDim.x + threadIdx.x;   // one float4 per thread
    float4 xv = ((const float4*)x)[i];
    float4 tv = ((const float4*)topo)[i];
    float e = (*is_end != 0) ? 1.0f : 0.0f;
    v4h a = { (_Float16)xv.x, (_Float16)xv.y, (_Float16)xv.z, (_Float16)xv.w };
    v4h b = { (_Float16)(xv.x + e * tv.x), (_Float16)(xv.y + e * tv.y),
              (_Float16)(xv.z + e * tv.z), (_Float16)(xv.w + e * tv.w) };
    ((v4h*)x16)[i]  = a;
    ((v4h*)xt16)[i] = b;
}

__global__ __launch_bounds__(256) void cvt_w_kernel(const float* __restrict__ src,
                                                    _Float16* __restrict__ dst) {
    size_t i = (size_t)blockIdx.x * blockDim.x + threadIdx.x;   // one float4 per thread
    float4 s = ((const float4*)src)[i];
    v4h d = { (_Float16)s.x, (_Float16)s.y, (_Float16)s.z, (_Float16)s.w };
    ((v4h*)dst)[i] = d;
}

// ---------------- generic 16x64-per-wave GEMM: C = A[M,512] * W[N,512]^T ----------------
// MODE 0: Q epilogue  -> Q16[b,h,n,d]              (Nout=512)
// MODE 1: KV epilogue -> K16[b,h,n,d], Vt16[b,h,d,n (stride SEQP)]   (Nout=1024)
// MODE 2: proj epilogue -> f32 d_out[t,f] + bias   (Nout=512)
template<int MODE>
__global__ __launch_bounds__(256) void gemm_ws(const _Float16* __restrict__ A,
                                               const _Float16* __restrict__ W,
                                               _Float16* __restrict__ outK,
                                               _Float16* __restrict__ outV,
                                               const float* __restrict__ bias,
                                               float* __restrict__ outF) {
    const int bm   = blockIdx.x % MBLOCKS;
    const int bn   = blockIdx.x / MBLOCKS;
    const int wave = threadIdx.x >> 5;
    const int lane = threadIdx.x & 31;
    const int mbase = bm * 128 + wave * 16;
    const int nbase = bn * 64;

    const _Float16* arow = A + (size_t)(mbase + (lane & 15)) * DIMX;

    v8f acc[4] = {};
    #pragma unroll 4
    for (int kb = 0; kb < 16; ++kb) {
        v16h a = load_a_frag(arow, kb, lane);
        #pragma unroll
        for (int ct = 0; ct < 4; ++ct) {
            const _Float16* wrow = W + (size_t)(nbase + ct * 16 + (lane & 15)) * DIMX;
            v16h b = load_b_frag(wrow, kb, lane);
            acc[ct] = wmma_f16(a, b, acc[ct]);
        }
    }

    #pragma unroll
    for (int ct = 0; ct < 4; ++ct) {
        const int f = nbase + ct * 16 + (lane & 15);
        #pragma unroll
        for (int v = 0; v < 8; ++v) {
            const int t = mbase + ((lane & 16) ? v + 8 : v);
            const float val = acc[ct][v];
            if (MODE == 2) {
                outF[(size_t)t * DIMX + f] = val + bias[f];
            } else {
                const int b = t / SEQ, n = t % SEQ;
                if (MODE == 0) {
                    const int h = f >> 6, d = f & 63;
                    outK[(((size_t)(b * HEADS + h)) * SEQ + n) * HD + d] = (_Float16)val;
                } else { // MODE == 1
                    const int which = f >> 9, f2 = f & 511;
                    const int h = f2 >> 6, d = f2 & 63;
                    if (which == 0)
                        outK[(((size_t)(b * HEADS + h)) * SEQ + n) * HD + d] = (_Float16)val;
                    else
                        outV[(((size_t)(b * HEADS + h)) * HD + d) * SEQP + n] = (_Float16)val;
                }
            }
        }
    }
}

// ---------------- fused attention: softmax(Q K^T * scale) V per (b,h) ----------------
__global__ __launch_bounds__(288) void attn_kernel(const _Float16* __restrict__ Q,
                                                   const _Float16* __restrict__ K,
                                                   const _Float16* __restrict__ Vt,
                                                   _Float16* __restrict__ O) {
    __shared__ _Float16 Plds[9][16][SEQP];    // 46080 bytes

    const int bh   = blockIdx.x;          // b*8 + h
    const int b    = bh >> 3;
    const int h    = bh & 7;
    const int wave = threadIdx.x >> 5;    // 0..8 == row tile
    const int lane = threadIdx.x & 31;

    const _Float16* Qb = Q  + (size_t)bh * SEQ * HD;
    const _Float16* Kb = K  + (size_t)bh * SEQ * HD;
    const _Float16* Vb = Vt + (size_t)bh * HD * SEQP;

    // Q fragments for this wave's 16 rows (K-dim = 64 -> 2 blocks), reused for all col tiles
    const _Float16* qrow = Qb + (size_t)(wave * 16 + (lane & 15)) * HD;
    const v16h aq0 = load_a_frag(qrow, 0, lane);
    const v16h aq1 = load_a_frag(qrow, 1, lane);

    // S = Q K^T over 9 column tiles
    v8f s[9];
    #pragma unroll
    for (int j = 0; j < 9; ++j) {
        const _Float16* krow = Kb + (size_t)(j * 16 + (lane & 15)) * HD;
        v8f acc = {};
        acc = wmma_f16(aq0, load_b_frag(krow, 0, lane), acc);
        acc = wmma_f16(aq1, load_b_frag(krow, 1, lane), acc);
        s[j] = acc;
    }

    // rowwise softmax: each row lives in 16 lanes (one half-wave) x 9 values
    const float SC = 0.125f;   // HEAD_DIM^-0.5
    #pragma unroll
    for (int v = 0; v < 8; ++v) {
        float m = -1e30f;
        #pragma unroll
        for (int j = 0; j < 9; ++j) { float q = s[j][v] * SC; s[j][v] = q; m = fmaxf(m, q); }
        m = fmaxf(m, __shfl_xor(m, 1));
        m = fmaxf(m, __shfl_xor(m, 2));
        m = fmaxf(m, __shfl_xor(m, 4));
        m = fmaxf(m, __shfl_xor(m, 8));
        float sum = 0.0f;
        #pragma unroll
        for (int j = 0; j < 9; ++j) { float e = __expf(s[j][v] - m); s[j][v] = e; sum += e; }
        sum += __shfl_xor(sum, 1);
        sum += __shfl_xor(sum, 2);
        sum += __shfl_xor(sum, 4);
        sum += __shfl_xor(sum, 8);
        const float inv = __frcp_rn(sum);
        const int rloc = (lane & 16) ? v + 8 : v;
        #pragma unroll
        for (int j = 0; j < 9; ++j)
            Plds[wave][rloc][j * 16 + (lane & 15)] = (_Float16)(s[j][v] * inv);
        Plds[wave][rloc][SEQ + (lane & 15)] = (_Float16)0.0f;   // zero pad cols 144..159
    }
    __syncthreads();

    // O = P[16 x 160] * V[160 x 64]  (pad rows of V killed by zero P)
    v8f o[4] = {};
    const _Float16* prow = &Plds[wave][lane & 15][0];
    #pragma unroll
    for (int kb = 0; kb < 5; ++kb) {
        v16h ap = load_a_frag(prow, kb, lane);        // ds_load_b128 x2
        #pragma unroll
        for (int ct = 0; ct < 4; ++ct) {
            const _Float16* vrow = Vb + (size_t)(ct * 16 + (lane & 15)) * SEQP;
            o[ct] = wmma_f16(ap, load_b_frag(vrow, kb, lane), o[ct]);
        }
    }

    // store O in [B, N, H*HD] token-major f16 layout for the projection GEMM
    #pragma unroll
    for (int ct = 0; ct < 4; ++ct) {
        #pragma unroll
        for (int v = 0; v < 8; ++v) {
            const int n = wave * 16 + ((lane & 16) ? v + 8 : v);
            O[((size_t)(b * SEQ + n)) * DIMX + h * HD + ct * 16 + (lane & 15)] = (_Float16)o[ct][v];
        }
    }
}

// ---------------- host-side launch ----------------
extern "C" void kernel_launch(void* const* d_in, const int* in_sizes, int n_in,
                              void* d_out, int out_size, void* d_ws, size_t ws_size,
                              hipStream_t stream) {
    const float* x      = (const float*)d_in[0];
    const float* topo   = (const float*)d_in[1];
    const float* kv_w   = (const float*)d_in[2];
    const float* q_w    = (const float*)d_in[3];
    const float* proj_w = (const float*)d_in[4];
    const float* proj_b = (const float*)d_in[5];
    const int*   is_end = (const int*)d_in[6];
    float* out = (float*)d_out;

    _Float16* ws = (_Float16*)d_ws;
    _Float16* kvw16 = ws;                              // 1024*512
    _Float16* qw16  = kvw16 + 1024 * 512;              // 512*512
    _Float16* pw16  = qw16  + 512 * 512;               // 512*512
    _Float16* x16   = pw16  + 512 * 512;               // NTOK*512
    _Float16* xt16  = x16   + (size_t)NTOK * DIMX;     // NTOK*512
    _Float16* Q16   = xt16  + (size_t)NTOK * DIMX;     // NTOK*512 ([B,H,N,D])
    _Float16* K16   = Q16   + (size_t)NTOK * DIMX;     // NTOK*512 ([B,H,N,D])
    _Float16* Vt16  = K16   + (size_t)NTOK * DIMX;     // B*H*HD*SEQP ([B,H,D,Npad])
    _Float16* O16   = x16;  // alias: x16 fully consumed by the Q GEMM before attention runs

    // 1) f32 -> f16 conversions
    cvt_w_kernel<<<(1024 * 512 / 4) / 256, 256, 0, stream>>>(kv_w, kvw16);
    cvt_w_kernel<<<(512 * 512 / 4) / 256, 256, 0, stream>>>(q_w, qw16);
    cvt_w_kernel<<<(512 * 512 / 4) / 256, 256, 0, stream>>>(proj_w, pw16);
    cvt_x_kernel<<<((size_t)NTOK * DIMX / 4) / 256, 256, 0, stream>>>(x, topo, is_end, x16, xt16);

    // 2) KV projection (Nout=1024) and Q projection (Nout=512)
    gemm_ws<1><<<MBLOCKS * (1024 / 64), 256, 0, stream>>>(xt16, kvw16, K16, Vt16, nullptr, nullptr);
    gemm_ws<0><<<MBLOCKS * (512 / 64),  256, 0, stream>>>(x16,  qw16,  Q16,  nullptr, nullptr, nullptr);

    // 3) fused attention per (b,h)
    attn_kernel<<<BATCH * HEADS, 288, 0, stream>>>(Q16, K16, Vt16, O16);

    // 4) output projection + bias -> f32 d_out
    gemm_ws<2><<<MBLOCKS * (512 / 64), 256, 0, stream>>>(O16, pw16, nullptr, nullptr, proj_b, out);
}
